// PIB_2886218023066
// MI455X (gfx1250) — matrix operand order repaired and checked by reading, not compiled, not run on hardware
//
#include <hip/hip_runtime.h>
#include <hip/hip_bf16.h>
#include <math.h>

typedef __attribute__((ext_vector_type(16))) _Float16 v16h;
typedef __attribute__((ext_vector_type(8)))  _Float16 v8h;
typedef __attribute__((ext_vector_type(2)))  _Float16 v2h;
typedef __attribute__((ext_vector_type(8)))  float    v8f;
typedef unsigned int u32x4 __attribute__((ext_vector_type(4)));
typedef int          i32x4 __attribute__((ext_vector_type(4)));
typedef int          i32x8 __attribute__((ext_vector_type(8)));

#define B_DIM 8
#define N_TOK 8192
#define M_TOT (B_DIM * N_TOK)   // 65536 tokens
#define XD    1024
#define H_DIM 512
#define Z_D   256
#define P_DIM 8
#define C_DIM 4
#define S_DIM 50
#define TOPKN 256

// ---- output layout (floats, concat in reference return order) ----
#define DEC_OFF   0                                   // (8,4)    = 32
#define MU_OFF    32                                  // (8,256)  = 2048
#define SG_OFF    (MU_OFF + P_DIM * Z_D)              // (8,256)  = 2048
#define ZTOPK_OFF (SG_OFF + P_DIM * Z_D)              // (8,256,256)
#define MUT_OFF   (ZTOPK_OFF + B_DIM * TOPKN * Z_D)   // (8,256)
#define SGT_OFF   (MUT_OFF + B_DIM * Z_D)             // (8,256)

// ---- workspace layout (bytes) ----
#define WS_ZPN  0u                         // 2048 f32 normalized proxies
#define WS_FEAT (8u * 1024u)               // 2048 int  top-k token indices
#define WS_ATT  (16u * 1024u)              // 65536*8 f32 (2 MB)
#define WS_BUFA (4u << 20)                 // 64 MB: h1 (f16) then z (f32)
#define WS_BUFB ((4u << 20) + (64u << 20)) // 64 MB: h2 (f16)

__device__ __forceinline__ float softplusf(float x) {
    return fmaxf(x, 0.f) + log1pf(expf(-fabsf(x)));
}

// Tensor-DMA: 2D tile load (f16) Global -> LDS via the TDM engine.
// D# per CDNA5 ISA: group0 = {count, lds_addr,
// global_addr, type=2}; group1 = {data_size=1(2B), tensor dims (huge; tiles
// never OOB), tile_dim0/1, tensor_dim0_stride}. groups 2/3 zero (2D tensor).
__device__ __forceinline__ void tdm_load_tile_f16(unsigned lds_addr,
                                                  unsigned long gaddr,
                                                  int tile_w, int tile_h,
                                                  unsigned long stride_units) {
    u32x4 g0;
    g0[0] = 1u;                                      // count=1, user mode
    g0[1] = lds_addr;                                // lds_addr [63:32]
    g0[2] = (unsigned)(gaddr & 0xffffffffu);         // global_addr lo
    g0[3] = (unsigned)((gaddr >> 32) & 0x1ffffffu) | (2u << 30);  // hi | type=2
    const unsigned TD = 0x40000000u;                 // huge tensor dims: no OOB
    i32x8 g1;
    g1[0] = (int)(1u << 16);                                  // data_size=2B
    g1[1] = (int)((TD & 0xffffu) << 16);                      // tensor_dim0 lo
    g1[2] = (int)(((TD >> 16) & 0xffffu) | ((TD & 0xffffu) << 16));
    g1[3] = (int)(((TD >> 16) & 0xffffu) | ((unsigned)tile_w << 16));
    g1[4] = (int)((unsigned)tile_h);                          // tile_dim1, dim2=0
    g1[5] = (int)(unsigned)(stride_units & 0xffffffffu);      // dim0_stride lo
    g1[6] = (int)(unsigned)((stride_units >> 32) & 0xffffu);  // stride hi
    g1[7] = 0;
    i32x4 z4 = {0, 0, 0, 0};
#if __clang_major__ >= 23
    i32x8 z8 = {0, 0, 0, 0, 0, 0, 0, 0};
    __builtin_amdgcn_tensor_load_to_lds(g0, g1, z4, z4, z8, 0);
#else
    __builtin_amdgcn_tensor_load_to_lds(g0, g1, z4, z4, 0);
#endif
}

// ---------------------------------------------------------------------------
// Kernel 1: proxies -> mu, sigma, z_proxy, zp_norm, decoder logits
// decoder_logits = mean_s((mu + sigma*eps_s) @ W + b) = z_proxy @ W + b
// ---------------------------------------------------------------------------
__global__ void proxy_kernel(const float* __restrict__ proxies,
                             const float* __restrict__ eps,
                             const float* __restrict__ decW,
                             const float* __restrict__ decb,
                             float* __restrict__ out,
                             float* __restrict__ zpn) {
    __shared__ float zp[P_DIM * Z_D];
    __shared__ float ns[P_DIM];
    const int t = threadIdx.x;
    for (int e = 0; e < 8; ++e) {
        int idx = t + e * 256;
        int p = idx >> 8, zz = idx & 255;
        float mu = proxies[p * (2 * Z_D) + zz];
        float sg = softplusf(proxies[p * (2 * Z_D) + Z_D + zz]);
        float es = 0.f;
        for (int s = 0; s < S_DIM; ++s) es += eps[((long)(p * S_DIM + s)) * Z_D + zz];
        zp[idx] = mu + sg * (es * (1.f / (float)S_DIM));
        out[MU_OFF + idx] = mu;
        out[SG_OFF + idx] = sg;
    }
    __syncthreads();
    if (t < P_DIM) {
        float s = 0.f;
        for (int j = 0; j < Z_D; ++j) { float v = zp[t * Z_D + j]; s += v * v; }
        ns[t] = fmaxf(sqrtf(s), 1e-12f);
    }
    __syncthreads();
    for (int e = 0; e < 8; ++e) {
        int idx = t + e * 256;
        zpn[idx] = zp[idx] / ns[idx >> 8];
    }
    if (t < P_DIM * C_DIM) {
        int p = t >> 2, c = t & 3;
        float acc = decb[c];
        for (int j = 0; j < Z_D; ++j) acc += zp[p * Z_D + j] * decW[j * C_DIM + c];
        out[DEC_OFF + t] = acc;
    }
}

// ---------------------------------------------------------------------------
// WMMA GEMM: C[M,N] = act(A[M,K] @ W[K,N] + bias)
// Block: 256 threads = 8 waves in 2(M) x 4(N); block tile 64 x 256, K step 32.
// A tile in LDS as [64][32] f16 (TDM DMA when A is f16, else float4 loads ->
// v8h ds_store_b128); W tile transposed in LDS as [256][32] f16 via clustered
// float4 loads + packed b32 stores. Next W tile prefetched.
// ---------------------------------------------------------------------------
template <bool A_HALF, bool A_TDM, bool RELU, bool OUT_HALF>
__global__ __launch_bounds__(256)
void gemm_kernel(const void* __restrict__ Ap, const float* __restrict__ W,
                 const float* __restrict__ bias, void* __restrict__ Op,
                 int K, int N) {
    __shared__ __align__(32) _Float16 As[64 * 32];
    __shared__ __align__(32) _Float16 Bs[256 * 32];
    const int tid  = threadIdx.x;
    const int lane = tid & 31;
    const int wave = tid >> 5;
    const int wm   = wave >> 2;   // 0..1  -> 32 rows each
    const int wn   = wave & 3;    // 0..3  -> 64 cols each
    const int h    = lane >> 4;   // half-wave select
    const int r    = lane & 15;
    const long m0  = (long)blockIdx.x * 64;
    const int  n0  = blockIdx.y * 256;

    v8f acc[2][4] = {};
    const float*    Af = (const float*)Ap;
    const _Float16* Ah = (const _Float16*)Ap;

    for (int kc = 0; kc < K; kc += 32) {
        if (A_TDM) {
            // TDM copies the 64x32 f16 A tile straight into LDS; only wave 0
            // issues it (TENSORcnt-tracked), other waves overlap on B tile.
            if (wave == 0) {
                unsigned long ga =
                    (unsigned long)(const void*)Ah + (m0 * (long)K + kc) * 2l;
                tdm_load_tile_f16((unsigned)(size_t)(void*)As, ga, 32, 64,
                                  (unsigned long)K);
            }
        } else {
            // 8 consecutive K elems / thread: 2x b128 loads -> 1x b128 store
            int row = tid >> 2, kk0 = (tid & 3) * 8;
            if (A_HALF) {
                v8h av = *(const v8h*)&Ah[(m0 + row) * (long)K + kc + kk0];
                *(v8h*)&As[row * 32 + kk0] = av;
            } else {
                const float4* src =
                    (const float4*)&Af[(m0 + row) * (long)K + kc + kk0];
                float4 f0 = src[0], f1 = src[1];
                v8h av = {(_Float16)f0.x, (_Float16)f0.y, (_Float16)f0.z,
                          (_Float16)f0.w, (_Float16)f1.x, (_Float16)f1.y,
                          (_Float16)f1.z, (_Float16)f1.w};
                *(v8h*)&As[row * 32 + kk0] = av;
            }
        }
        // B tile: 4 cells/thread, each cell = 2 kk x 4 nn. Load phase first
        // (8 clustered b128 loads), then cvt_pk + transposed b32 LDS stores.
        float4 wv[8];
        #pragma unroll
        for (int g = 0; g < 4; ++g) {
            int q = tid + g * 256;           // 1024 cells
            int kk2 = q >> 6;                // 0..15  (kk = 2*kk2)
            int nn4 = q & 63;                // nn = 4*nn4
            const float* base = &W[(long)(kc + 2 * kk2) * N + n0 + 4 * nn4];
            wv[2 * g]     = *(const float4*)base;
            wv[2 * g + 1] = *(const float4*)(base + N);
        }
        #pragma unroll
        for (int g = 0; g < 4; ++g) {
            int q = tid + g * 256;
            int kk = (q >> 6) * 2;
            int nn = (q & 63) * 4;
            float4 f0 = wv[2 * g], f1 = wv[2 * g + 1];
            float a0[4] = {f0.x, f0.y, f0.z, f0.w};
            float a1[4] = {f1.x, f1.y, f1.z, f1.w};
            #pragma unroll
            for (int c = 0; c < 4; ++c) {
                v2h pk = {(_Float16)a0[c], (_Float16)a1[c]};  // (kk, kk+1)
                *(v2h*)&Bs[(nn + c) * 32 + kk] = pk;
            }
        }
        if (kc + 32 < K) {                   // prefetch next W tile
            __builtin_prefetch(
                &W[(long)(kc + 32 + (tid & 31)) * N + n0 + (tid >> 5) * 32], 0, 0);
        }
        if (A_TDM) {
            if (wave == 0) __builtin_amdgcn_s_wait_tensorcnt(0);
        }
        __syncthreads();

        v16h afr[2], bfr[4];
        #pragma unroll
        for (int mi = 0; mi < 2; ++mi) {
            // A frag: row striped in lane; k = 16h..16h+15 contiguous
            const _Float16* rp = &As[(wm * 32 + mi * 16 + r) * 32];
            v8h lo = *(const v8h*)(rp + 16 * h);
            v8h hi = *(const v8h*)(rp + 16 * h + 8);
            afr[mi] = __builtin_shufflevector(lo, hi, 0, 1, 2, 3, 4, 5, 6, 7,
                                              8, 9, 10, 11, 12, 13, 14, 15);
        }
        #pragma unroll
        for (int ni = 0; ni < 4; ++ni) {
            // B frag: col per lane; k = 16h..16h+15 contiguous (transposed LDS)
            bfr[ni] = *(const v16h*)&Bs[(wn * 64 + ni * 16 + r) * 32 + 16 * h];
        }
        #pragma unroll
        for (int mi = 0; mi < 2; ++mi)
            #pragma unroll
            for (int ni = 0; ni < 4; ++ni)
                acc[mi][ni] = __builtin_amdgcn_wmma_f32_16x16x32_f16(
                    false, afr[mi], false, bfr[ni], (short)0, acc[mi][ni],
                    false, false);
        __syncthreads();
    }
    // epilogue: C/D element e -> row (e + 8h), col (lane&15) within 16x16 tile
    #pragma unroll
    for (int mi = 0; mi < 2; ++mi) {
        #pragma unroll
        for (int ni = 0; ni < 4; ++ni) {
            int col = n0 + wn * 64 + ni * 16 + r;
            float bv = bias[col];
            #pragma unroll
            for (int e = 0; e < 8; ++e) {
                long grow = m0 + wm * 32 + mi * 16 + h * 8 + e;
                float v = acc[mi][ni][e] + bv;
                if (RELU) v = fmaxf(v, 0.f);
                if (OUT_HALF) ((_Float16*)Op)[grow * N + col] = (_Float16)v;
                else          ((float*)Op)[grow * N + col] = v;
            }
        }
    }
}

// ---------------------------------------------------------------------------
// att[m][p] = dot(z[m], zp_norm[p]) / max(||z[m]||, 1e-12)
// ---------------------------------------------------------------------------
__global__ void att_kernel(const float* __restrict__ z,
                           const float* __restrict__ zpn,
                           float* __restrict__ att) {
    __shared__ float zp[P_DIM * Z_D];
    const int t = threadIdx.x;
    for (int e = 0; e < 8; ++e) zp[t + e * 256] = zpn[t + e * 256];
    __syncthreads();
    long m = (long)blockIdx.x * 256 + t;
    const float4* zr = (const float4*)(z + m * Z_D);
    float dot[P_DIM];
    #pragma unroll
    for (int p = 0; p < P_DIM; ++p) dot[p] = 0.f;
    float sq = 0.f;
    for (int j4 = 0; j4 < Z_D / 4; ++j4) {
        float4 v = zr[j4];
        float vv[4] = {v.x, v.y, v.z, v.w};
        #pragma unroll
        for (int c = 0; c < 4; ++c) {
            float zv = vv[c];
            sq += zv * zv;
            #pragma unroll
            for (int p = 0; p < P_DIM; ++p) dot[p] += zv * zp[p * Z_D + j4 * 4 + c];
        }
    }
    float inv = 1.f / fmaxf(sqrtf(sq), 1e-12f);
    #pragma unroll
    for (int p = 0; p < P_DIM; ++p) att[m * P_DIM + p] = dot[p] * inv;
}

// ---------------------------------------------------------------------------
// Per-batch top-256 selection over att[b] (65536 values) in LDS (256KB dyn).
// One ranking serves both flattenings: idx = n*P+p gives (p -> counts,
// n -> z_topk gather order). Tie-break: lower flat index wins (top_k rule).
// ---------------------------------------------------------------------------
__global__ void topk_kernel(float* __restrict__ att,
                            const float* __restrict__ proxies,
                            int* __restrict__ feat, float* __restrict__ out) {
    extern __shared__ float sAtt[];
    __shared__ float sval[256];
    __shared__ int   sidx[256];
    __shared__ int   counts[P_DIM];
    __shared__ int   sPos;
    const int t = threadIdx.x, b = blockIdx.x;
    float* ab = att + (long)b * N_TOK * P_DIM;
    for (int i = t; i < N_TOK * P_DIM; i += 256) sAtt[i] = ab[i];
    if (t < P_DIM) counts[t] = 0;
    __syncthreads();
    for (int k = 0; k < TOPKN; ++k) {
        float best = -1e30f; int bi = 0x7fffffff;
        for (int i = t; i < N_TOK * P_DIM; i += 256) {
            float v = sAtt[i];
            if (v > best) { best = v; bi = i; }  // increasing i => tie keeps lowest
        }
        sval[t] = best; sidx[t] = bi;
        __syncthreads();
        for (int s = 128; s > 0; s >>= 1) {
            if (t < s) {
                float v2 = sval[t + s]; int i2 = sidx[t + s];
                if (v2 > sval[t] || (v2 == sval[t] && i2 < sidx[t])) {
                    sval[t] = v2; sidx[t] = i2;
                }
            }
            __syncthreads();
        }
        if (t == 0) {
            int idx = sidx[0];
            feat[b * TOPKN + k] = idx >> 3;   // n
            counts[idx & 7]++;                // p
            sAtt[idx] = -1e30f;
        }
        __syncthreads();
    }
    if (t == 0) {
        int bp = 0, bc = counts[0];
        for (int p = 1; p < P_DIM; ++p)
            if (counts[p] > bc) { bc = counts[p]; bp = p; }  // argmax, first max
        sPos = bp;
    }
    __syncthreads();
    int pos = sPos;
    out[MUT_OFF + b * Z_D + t] = proxies[pos * (2 * Z_D) + t];
    out[SGT_OFF + b * Z_D + t] = softplusf(proxies[pos * (2 * Z_D) + Z_D + t]);
}

// ---------------------------------------------------------------------------
// z_topk[b][k][:] = z[b][feat[b][k]][:]
// ---------------------------------------------------------------------------
__global__ void gather_kernel(const float* __restrict__ z,
                              const int* __restrict__ feat,
                              float* __restrict__ out) {
    int bk = blockIdx.x;
    int b = bk >> 8;
    int t = threadIdx.x;
    int n = feat[bk];
    out[ZTOPK_OFF + (long)bk * Z_D + t] = z[((long)b * N_TOK + n) * Z_D + t];
}

extern "C" void kernel_launch(void* const* d_in, const int* in_sizes, int n_in,
                              void* d_out, int out_size, void* d_ws, size_t ws_size,
                              hipStream_t stream) {
    (void)in_sizes; (void)n_in; (void)out_size; (void)ws_size;
    const float* x       = (const float*)d_in[0];
    const float* W1      = (const float*)d_in[1];
    const float* b1      = (const float*)d_in[2];
    const float* W2      = (const float*)d_in[3];
    const float* b2      = (const float*)d_in[4];
    const float* W3      = (const float*)d_in[5];
    const float* b3      = (const float*)d_in[6];
    const float* proxies = (const float*)d_in[7];
    const float* eps     = (const float*)d_in[8];
    const float* decW    = (const float*)d_in[9];
    const float* decb    = (const float*)d_in[10];
    float* out = (float*)d_out;
    char*  ws  = (char*)d_ws;

    float* zpn  = (float*)(ws + WS_ZPN);
    int*   feat = (int*)(ws + WS_FEAT);
    float* att  = (float*)(ws + WS_ATT);
    void*  bufA = (void*)(ws + WS_BUFA);  // h1 (f16) then z (f32), aliased
    void*  bufB = (void*)(ws + WS_BUFB);  // h2 (f16)

    proxy_kernel<<<1, 256, 0, stream>>>(proxies, eps, decW, decb, out, zpn);

    gemm_kernel<false, false, true, true>
        <<<dim3(M_TOT / 64, H_DIM / 256), 256, 0, stream>>>(
            x, W1, b1, bufA, XD, H_DIM);
    gemm_kernel<true, true, true, true>
        <<<dim3(M_TOT / 64, H_DIM / 256), 256, 0, stream>>>(
            bufA, W2, b2, bufB, H_DIM, H_DIM);
    gemm_kernel<true, true, false, false>
        <<<dim3(M_TOT / 64, Z_D / 256), 256, 0, stream>>>(
            bufB, W3, b3, bufA, H_DIM, Z_D);

    att_kernel<<<M_TOT / 256, 256, 0, stream>>>((const float*)bufA, zpn, att);
    topk_kernel<<<B_DIM, 256, N_TOK * P_DIM * sizeof(float), stream>>>(
        att, proxies, feat, out);
    gather_kernel<<<B_DIM * TOPKN, 256, 0, stream>>>((const float*)bufA, feat, out);
}
